// Retrieval_RIM_80985903333780
// MI455X (gfx1250) — compile-verified, matrix operand order checked by reading, not run on hardware
//
// Retrieval-RIM pipeline for MI455X (gfx1250), wave32 + WMMA bf16 + TDM.
//
// Heavy GEMMs (token MLP 128->256 over 91.5k rows, RIM gate GEMMs) run on
// v_wmma_f32_16x16x32_bf16 with f32 accumulation.  Weights are pre-swizzled
// once per workgroup into LDS in WMMA B-fragment order (packed bf16); the
// inner loop is software-pipelined with ping-pong fragment buffers (no
// register rotation) so ds_load_b128 overlaps the XDL pipe.  Activation
// tiles are DMA'd into LDS with the Tensor Data Mover (tensor_load_to_lds +
// s_wait_tensorcnt).  All tanh use the native V_TANH_F32 transcendental.
#include <hip/hip_runtime.h>

typedef __bf16 bf16_t;
typedef bf16_t   v16bf __attribute__((ext_vector_type(16)));
typedef float    v8f   __attribute__((ext_vector_type(8)));
typedef unsigned v4u   __attribute__((ext_vector_type(4)));
typedef unsigned v8u   __attribute__((ext_vector_type(8)));

__device__ __forceinline__ bf16_t f2bf(float x) { return (bf16_t)x; }
__device__ __forceinline__ unsigned bf16_bits(float x) {
  return (unsigned)__builtin_bit_cast(unsigned short, (bf16_t)x);
}
__device__ __forceinline__ float sigm(float x) { return 1.f / (1.f + __expf(-x)); }

// Native CDNA5 tanh (V_TANH_F32, TRANS32) with a branch-free fallback.
__device__ __forceinline__ float fast_tanh(float x) {
#if __has_builtin(__builtin_amdgcn_tanhf)
  return __builtin_amdgcn_tanhf(x);
#elif __has_builtin(__builtin_amdgcn_tanh_f32)
  return __builtin_amdgcn_tanh_f32(x);
#else
  const float e = __expf(2.f * x);
  return 1.f - 2.f / (e + 1.f);
#endif
}

// ---------------------------------------------------------------------------
// Tensor Data Mover: DMA a 2D tile (dim0 elems/row x dim1 rows, f32, row
// stride stride0 elems) from global memory into LDS (contiguous), per the
// D# layout in CDNA5 ISA ch.8.  Issued once per wave; tracked by TENSORcnt.
// ---------------------------------------------------------------------------
__device__ __forceinline__ void tdm_load_2d_to_lds(const void* gsrc,
                                                   unsigned lds_byte_off,
                                                   unsigned dim0, unsigned dim1,
                                                   unsigned stride0) {
  const unsigned long long ga = (unsigned long long)gsrc;
  v4u g0; v8u g1;
  g0[0] = 1u;                                            // count=1, user mode
  g0[1] = lds_byte_off;                                  // D#.lds_addr
  g0[2] = (unsigned)ga;                                  // global_addr[31:0]
  g0[3] = ((unsigned)(ga >> 32) & 0x01FFFFFFu) | (2u << 30);  // addr[56:32]|type=2
  g1[0] = 2u << 16;                                      // data_size = 4 bytes
  g1[1] = (dim0 & 0xFFFFu) << 16;                        // tensor_dim0 lo16
  g1[2] = (dim0 >> 16) | ((dim1 & 0xFFFFu) << 16);       // dim0 hi16 | dim1 lo16
  g1[3] = (dim1 >> 16) | (dim0 << 16);                   // dim1 hi16 | tile_dim0
  g1[4] = dim1 & 0xFFFFu;                                // tile_dim1 (tile_dim2=0)
  g1[5] = stride0;                                       // tensor_dim0_stride lo32
  g1[6] = 0u;                                            // stride0 hi16 | stride1 lo16
  g1[7] = 0u;                                            // stride1 hi32
  asm volatile("tensor_load_to_lds %0, %1" :: "s"(g0), "s"(g1) : "memory");
}

// ---------------------------------------------------------------------------
// WMMA fragment helpers (CDNA5 ISA 7.12.2, 16-bit 16x32 layout).
// Lane l: rows/cols = l&15; K-half base = (l>>4)*8.
// VGPR v<4 packs K = base+2v(+1); v>=4 packs K = 16+base+2(v-4)(+1).
// ---------------------------------------------------------------------------
__device__ __forceinline__ v16bf load_a_frag(const float* src, int ld, int k0, int lane) {
  const int row = lane & 15;
  const int kb  = (lane >> 4) * 8;
  v16bf a;
#pragma unroll
  for (int v = 0; v < 8; ++v) {
    const int k = (v < 4) ? (kb + 2 * v) : (16 + kb + 2 * (v - 4));
    const float* p = src + (size_t)row * ld + k0 + k;
    a[2 * v]     = f2bf(p[0]);
    a[2 * v + 1] = f2bf(p[1]);
  }
  return a;
}

// Pre-swizzle a (K x N) row-major f32 weight block into LDS as packed-bf16
// B-fragments: frag f = ntile*(K/32)+kchunk, 32 lanes x 8 dwords each.
__device__ void stage_w_lds(unsigned* lds, const float* __restrict__ W, int ldn,
                            int K, int Nn, int tid, int nthreads) {
  const int nk = K >> 5, nt = Nn >> 4;
  const int total = nt * nk * 32 * 8;
  for (int idx = tid; idx < total; idx += nthreads) {
    const int w    = idx & 7;
    const int lane = (idx >> 3) & 31;
    const int rem  = idx >> 8;
    const int kk   = rem % nk;
    const int t    = rem / nk;
    const int col  = t * 16 + (lane & 15);
    const int kb   = (lane >> 4) * 8;
    const int k    = kk * 32 + ((w < 4) ? (kb + 2 * w) : (16 + kb + 2 * (w - 4)));
    lds[idx] = bf16_bits(W[(size_t)k * ldn + col]) |
               (bf16_bits(W[(size_t)(k + 1) * ldn + col]) << 16);
  }
}

__device__ __forceinline__ v16bf load_b_lds(const unsigned* lds, int frag, int lane) {
  const unsigned* p = lds + ((size_t)frag * 32 + lane) * 8;
  v8u u;
#pragma unroll
  for (int w = 0; w < 8; ++w) u[w] = p[w];
  return __builtin_bit_cast(v16bf, u);
}

#define WMMA_BF16(A, B, C) \
  __builtin_amdgcn_wmma_f32_16x16x32_bf16(false, (A), false, (B), (short)0, (C), false, false)

// ---------------------------------------------------------------------------
// Top-k helpers (register insertion list, descending values; ties -> first).
// ---------------------------------------------------------------------------
template <int K>
__device__ __forceinline__ void topk_insert(float (&tv)[K], int (&ti)[K], float v, int idx) {
  if (v > tv[K - 1]) {
    int p = K - 1;
    while (p > 0 && v > tv[p - 1]) { tv[p] = tv[p - 1]; ti[p] = ti[p - 1]; --p; }
    tv[p] = v; ti[p] = idx;
  }
}
template <int K>
__device__ __forceinline__ void sort_idx_asc(int (&ti)[K]) {
  for (int a = 1; a < K; ++a) {
    int key = ti[a], c = a - 1;
    while (c >= 0 && ti[c] > key) { ti[c + 1] = ti[c]; --c; }
    ti[c + 1] = key;
  }
}

// ---------------------------------------------------------------------------
// Kernel 1: fused attention-MLP scorer.
// score[r] = tanh( dot( tanh(row_r @ W1 + b1), W2 ) + b2 ), row = 128 f32.
// One wave per 16-row tile.  A tile arrives via TDM DMA into this wave's LDS
// slice while all 256 threads cooperatively swizzle W1 (+b1,W2) into LDS.
// The 16 N-tile loop is unrolled x2 with ping-pong B-fragment buffers so the
// prefetch ds-loads overlap the XDL pipe with no register rotation.
// ---------------------------------------------------------------------------
__global__ __launch_bounds__(256) void mlp_score_kernel(
    const float* __restrict__ rows, int nrows,
    const float* __restrict__ W1,   // (128,256) row-major
    const float* __restrict__ b1,   // (256)
    const float* __restrict__ W2,   // (256)
    const float* __restrict__ b2,   // (1)
    float* __restrict__ scores) {
  const int tid  = threadIdx.x;
  const int lane = tid & 31;
  const int wave = tid >> 5;
  const int tile = blockIdx.x * 8 + wave;
  const bool active = (tile * 16 < nrows);

  __shared__ unsigned w1_lds[16 * 4 * 32 * 8];  // 64 KB swizzled bf16 W1
  __shared__ float    a_lds[8 * 16 * 128];      // 64 KB: 8 per-wave A tiles
  __shared__ float    b1_lds[256];
  __shared__ float    w2_lds[256];

  if (active) {
    // DMA this wave's 16x128 f32 tile into LDS (contiguous rows -> 1 tile).
    tdm_load_2d_to_lds(rows + (size_t)tile * 2048,
                       (unsigned)(size_t)&a_lds[wave * 2048], 128u, 16u, 128u);
  }
  stage_w_lds(w1_lds, W1, 256, 128, 256, tid, 256);
  b1_lds[tid] = b1[tid];
  w2_lds[tid] = W2[tid];
  __syncthreads();
  if (!active) return;

  __builtin_amdgcn_s_wait_tensorcnt(0);
  const float* A = &a_lds[wave * 2048];
  v16bf a0 = load_a_frag(A, 128, 0, lane);
  v16bf a1 = load_a_frag(A, 128, 32, lane);
  v16bf a2 = load_a_frag(A, 128, 64, lane);
  v16bf a3 = load_a_frag(A, 128, 96, lane);

  float pd[8];
#pragma unroll
  for (int j = 0; j < 8; ++j) pd[j] = 0.f;

  v16bf bA[4], bB[4];
#pragma unroll
  for (int kk = 0; kk < 4; ++kk) bA[kk] = load_b_lds(w1_lds, kk, lane);

  for (int nt = 0; nt < 16; nt += 2) {
    // even tile: compute from bA, prefetch tile nt+1 into bB
#pragma unroll
    for (int kk = 0; kk < 4; ++kk)
      bB[kk] = load_b_lds(w1_lds, (nt + 1) * 4 + kk, lane);
    {
      v8f c = {};
      c = WMMA_BF16(a0, bA[0], c);
      c = WMMA_BF16(a1, bA[1], c);
      c = WMMA_BF16(a2, bA[2], c);
      c = WMMA_BF16(a3, bA[3], c);
      const int col = nt * 16 + (lane & 15);
      const float bb = b1_lds[col];
      const float w2 = w2_lds[col];
#pragma unroll
      for (int j = 0; j < 8; ++j) pd[j] += fast_tanh(c[j] + bb) * w2;
    }
    // odd tile: compute from bB, prefetch tile nt+2 into bA
    if (nt + 2 < 16) {
#pragma unroll
      for (int kk = 0; kk < 4; ++kk)
        bA[kk] = load_b_lds(w1_lds, (nt + 2) * 4 + kk, lane);
    }
    {
      v8f c = {};
      c = WMMA_BF16(a0, bB[0], c);
      c = WMMA_BF16(a1, bB[1], c);
      c = WMMA_BF16(a2, bB[2], c);
      c = WMMA_BF16(a3, bB[3], c);
      const int col = (nt + 1) * 16 + (lane & 15);
      const float bb = b1_lds[col];
      const float w2 = w2_lds[col];
#pragma unroll
      for (int j = 0; j < 8; ++j) pd[j] += fast_tanh(c[j] + bb) * w2;
    }
  }
  // Reduce the 16 N-lanes sharing each row (C layout: row = j + 8*(lane>=16)).
#pragma unroll
  for (int j = 0; j < 8; ++j) {
    float v = pd[j];
    v += __shfl_xor(v, 1, 32);
    v += __shfl_xor(v, 2, 32);
    v += __shfl_xor(v, 4, 32);
    v += __shfl_xor(v, 8, 32);
    pd[j] = v;
  }
  if ((lane & 15) == 0) {
    const int rbase = tile * 16 + ((lane >> 4) ? 8 : 0);
    const float bias2 = b2[0];
#pragma unroll
    for (int j = 0; j < 8; ++j) scores[rbase + j] = fast_tanh(pd[j] + bias2);
  }
}

// ---------------------------------------------------------------------------
// Kernel 2: per-row softmax over T plus weighted pooling over tokens.
// ---------------------------------------------------------------------------
__global__ __launch_bounds__(256) void softmax_pool_kernel(
    const float* __restrict__ tok, const float* __restrict__ scores,
    float* __restrict__ out_w, float* __restrict__ pooled, int nrows, int T) {
  const int lane = threadIdx.x & 31;
  const int r = blockIdx.x * (blockDim.x >> 5) + (threadIdx.x >> 5);
  if (r >= nrows) return;
  float s = (lane < T) ? scores[(size_t)r * T + lane] : -3.4e38f;
  float m = s;
  for (int off = 16; off; off >>= 1) m = fmaxf(m, __shfl_xor(m, off, 32));
  float e = (lane < T) ? __expf(s - m) : 0.f;
  float sum = e;
  for (int off = 16; off; off >>= 1) sum += __shfl_xor(sum, off, 32);
  const float w = e / sum;
  if (lane < T) out_w[(size_t)r * T + lane] = w;
  const float* base = tok + (size_t)r * T * 128;
#pragma unroll
  for (int dc = 0; dc < 4; ++dc) {
    const int d = lane + 32 * dc;
    float acc = 0.f;
    for (int t = 0; t < T; ++t) acc += __shfl(w, t, 32) * base[(size_t)t * 128 + d];
    pooled[(size_t)r * 128 + d] = acc;
  }
}

// ---------------------------------------------------------------------------
// Kernel 3: event mask + softmax over S (per batch), plus exact top-4 idx.
// ---------------------------------------------------------------------------
__global__ void event_softmax_kernel(const float* __restrict__ wev,
                                     const unsigned char* __restrict__ mask,
                                     float* __restrict__ qw,
                                     float* __restrict__ st_idx_out) {
  const int b = blockIdx.x, lane = threadIdx.x;
  float s = -3.4e38f;
  if (lane < 30) s = mask[b * 30 + lane] ? -9.0e15f : wev[b * 30 + lane];
  float m = s;
  for (int off = 16; off; off >>= 1) m = fmaxf(m, __shfl_xor(m, off, 32));
  float e = (lane < 30) ? __expf(s - m) : 0.f;
  float sum = e;
  for (int off = 16; off; off >>= 1) sum += __shfl_xor(sum, off, 32);
  const float w = e / sum;
  if (lane < 30) qw[b * 30 + lane] = w;
  float tv[4] = {-3.4e38f, -3.4e38f, -3.4e38f, -3.4e38f};
  int   ti[4] = {0, 0, 0, 0};
  for (int s2 = 0; s2 < 30; ++s2) {
    const float v = __shfl(w, s2, 32);  // all lanes execute
    topk_insert<4>(tv, ti, v, s2);
  }
  if (lane == 0) {
    sort_idx_asc<4>(ti);
#pragma unroll
    for (int k = 0; k < 4; ++k) st_idx_out[b * 4 + k] = (float)ti[k];
  }
}

// ---------------------------------------------------------------------------
// Kernel 4: perturbed top-4 over S=30 (short-term), LDS ds_add_f32 counts.
// ---------------------------------------------------------------------------
__global__ __launch_bounds__(256) void ptopk_st_kernel(
    const float* __restrict__ qw, const float* __restrict__ noise,
    const float* __restrict__ sigma_p, float* __restrict__ st_ind, int N) {
  const int b = blockIdx.x, tid = threadIdx.x;
  const float sigma = sigma_p[0];
  __shared__ float q_s[30];
  __shared__ float counts[4 * 30];
  if (tid < 30) q_s[tid] = qw[b * 30 + tid];
  for (int i = tid; i < 4 * 30; i += blockDim.x) counts[i] = 0.f;
  __syncthreads();
  for (int n = tid; n < N; n += blockDim.x) {
    const float* nr = noise + ((size_t)b * N + n) * 30;
    float tv[4] = {-3.4e38f, -3.4e38f, -3.4e38f, -3.4e38f};
    int   ti[4] = {0, 0, 0, 0};
    for (int s = 0; s < 30; ++s) topk_insert<4>(tv, ti, q_s[s] + nr[s] * sigma, s);
    sort_idx_asc<4>(ti);
#pragma unroll
    for (int k = 0; k < 4; ++k) atomicAdd(&counts[k * 30 + ti[k]], 1.f);
  }
  __syncthreads();
  const float inv = 1.f / (float)N;
  for (int i = tid; i < 4 * 30; i += blockDim.x)
    st_ind[(size_t)b * 4 * 30 + i] = counts[i] * inv;
}

// Kernel 5: new_query[b,q,:] = sum_s st_ind[b,q,s] * short[b,s,:]
__global__ void new_query_kernel(const float* __restrict__ st_ind,
                                 const float* __restrict__ shortp,
                                 float* __restrict__ nq) {
  const int b = blockIdx.x >> 2, q = blockIdx.x & 3, d = threadIdx.x;
  float acc = 0.f;
  for (int s = 0; s < 30; ++s)
    acc += st_ind[((size_t)b * 4 + q) * 30 + s] * shortp[((size_t)b * 30 + s) * 128 + d];
  nq[((size_t)b * 4 + q) * 128 + d] = acc;
}

// Kernel 6: tvec[i,b,:] = bilinear[i] @ q[b,i,:]
__global__ void bilinear_t_kernel(const float* __restrict__ bil,
                                  const float* __restrict__ nq,
                                  float* __restrict__ tvec) {
  const int i = blockIdx.x >> 4, b = blockIdx.x & 15, d = threadIdx.x;
  const float* w = bil + ((size_t)i * 128 + d) * 128;
  const float* q = nq + ((size_t)b * 4 + i) * 128;
  float acc = 0.f;
  for (int e = 0; e < 128; ++e) acc += w[e] * q[e];
  tvec[((size_t)i * 16 + b) * 128 + d] = acc;
}

// Kernel 7: sim + mask + softmax over M=256.  wsoft: (Q,B,256).
__global__ __launch_bounds__(256) void sim_softmax_kernel(
    const float* __restrict__ mem_e, const float* __restrict__ tvec,
    const unsigned char* __restrict__ mask, float* __restrict__ wsoft) {
  const int i = blockIdx.x >> 4, b = blockIdx.x & 15, m = threadIdx.x;
  const float* me = mem_e + ((size_t)b * 256 + m) * 128;
  const float* t  = tvec + ((size_t)i * 16 + b) * 128;
  float acc = 0.f;
#pragma unroll 4
  for (int d = 0; d < 128; ++d) acc += me[d] * t[d];
  if (mask[b * 256 + m]) acc = -9.0e15f;
  __shared__ float sc[256];
  sc[m] = acc; __syncthreads();
  for (int off = 128; off; off >>= 1) { if (m < off) sc[m] = fmaxf(sc[m], sc[m + off]); __syncthreads(); }
  const float mx = sc[0]; __syncthreads();
  const float e = __expf(acc - mx);
  sc[m] = e; __syncthreads();
  for (int off = 128; off; off >>= 1) { if (m < off) sc[m] += sc[m + off]; __syncthreads(); }
  wsoft[((size_t)i * 16 + b) * 256 + m] = e / sc[0];
}

// ---------------------------------------------------------------------------
// Kernel 8: perturbed top-8 over M=256 (long-term), per (i,b) block.
// ---------------------------------------------------------------------------
__global__ __launch_bounds__(256) void ptopk_lt_kernel(
    const float* __restrict__ wsoft, const float* __restrict__ noise,
    const float* __restrict__ sigma_p, float* __restrict__ lt_ind,
    float* __restrict__ lt_idx_out, int N) {
  const int i = blockIdx.x >> 4, b = blockIdx.x & 15, tid = threadIdx.x;
  const float sigma = sigma_p[0];
  __shared__ float wrow[256];
  __shared__ float counts[8 * 256];
  wrow[tid] = wsoft[((size_t)i * 16 + b) * 256 + tid];
  for (int j = tid; j < 8 * 256; j += blockDim.x) counts[j] = 0.f;
  __syncthreads();
  for (int n = tid; n < N; n += blockDim.x) {
    const float* nr = noise + (((size_t)i * 16 + b) * N + n) * 256;
    float tv[8]; int ti[8];
#pragma unroll
    for (int k = 0; k < 8; ++k) { tv[k] = -3.4e38f; ti[k] = 0; }
    for (int m = 0; m < 256; ++m) topk_insert<8>(tv, ti, wrow[m] + nr[m] * sigma, m);
    sort_idx_asc<8>(ti);
#pragma unroll
    for (int k = 0; k < 8; ++k) atomicAdd(&counts[k * 256 + ti[k]], 1.f);
  }
  __syncthreads();
  const float inv = 1.f / (float)N;
  for (int j = tid; j < 8 * 256; j += blockDim.x)
    lt_ind[((size_t)i * 16 + b) * 8 * 256 + j] = counts[j] * inv;
  if (tid == 0) {
    float tv[8]; int ti[8];
#pragma unroll
    for (int k = 0; k < 8; ++k) { tv[k] = -3.4e38f; ti[k] = 0; }
    for (int m = 0; m < 256; ++m) topk_insert<8>(tv, ti, wrow[m], m);
    sort_idx_asc<8>(ti);
#pragma unroll
    for (int k = 0; k < 8; ++k) lt_idx_out[((size_t)b * 4 + i) * 8 + k] = (float)ti[k];
  }
}

// Kernel 9: new_mem[i,b,k,:] = sum_m lt_ind[i,b,k,m] * mem_e[b,m,:]
__global__ void new_mem_kernel(const float* __restrict__ lt_ind,
                               const float* __restrict__ mem_e,
                               float* __restrict__ nmem) {
  const int k = blockIdx.x & 7, ib = blockIdx.x >> 3;
  const int i = ib >> 4, b = ib & 15, d = threadIdx.x;
  const float* li = lt_ind + (((size_t)i * 16 + b) * 8 + k) * 256;
  float acc = 0.f;
  for (int m = 0; m < 256; ++m) acc += li[m] * mem_e[((size_t)b * 256 + m) * 128 + d];
  nmem[(((size_t)i * 16 + b) * 8 + k) * 128 + d] = acc;
}

// ---------------------------------------------------------------------------
// Kernel 10: RIM LSTM chains.  One workgroup per (sequence, layer l); the
// batch B=16 is one WMMA M-tile.  Wave w owns gate cols [64w, 64w+64).
// Wx/Wh are swizzled into LDS as bf16 once and reused across all timesteps;
// each step's x_t batch (16 rows, strided) is gathered by a TDM 2D descriptor.
// ---------------------------------------------------------------------------
__global__ __launch_bounds__(128) void rim_kernel(
    const float* __restrict__ shortp, const float* __restrict__ nq,
    const float* __restrict__ nmem,
    const float* __restrict__ rh_Wx, const float* __restrict__ rh_Wh, const float* __restrict__ rh_b,
    const float* __restrict__ rt_Wx, const float* __restrict__ rt_Wh, const float* __restrict__ rt_b,
    const float* __restrict__ h0, const float* __restrict__ c0,
    float* __restrict__ feats) {
  const int seq = blockIdx.x >> 2;
  const int l   = blockIdx.x & 3;
  const int tid = threadIdx.x;
  const int lane = tid & 31, wave = tid >> 5;
  const bool today = (seq == 4);
  const int Tlen = today ? 30 : 9;
  const float* Wx = (today ? rt_Wx : rh_Wx) + (size_t)l * 128 * 256;
  const float* Wh = (today ? rt_Wh : rh_Wh) + (size_t)l * 64 * 256;
  const float* bs = (today ? rt_b : rh_b) + (size_t)l * 256;

  __shared__ unsigned wx_lds[16 * 4 * 32 * 8];  // 64 KB swizzled bf16 Wx
  __shared__ unsigned wh_lds[16 * 2 * 32 * 8];  // 32 KB swizzled bf16 Wh
  __shared__ float bias_s[256];
  __shared__ float xt_s[16 * 128];
  __shared__ float h_s[16 * 64];
  __shared__ float c_s[16 * 64];
  __shared__ float g_s[16 * 256];

  stage_w_lds(wx_lds, Wx, 256, 128, 256, tid, 128);
  stage_w_lds(wh_lds, Wh, 256, 64, 256, tid, 128);
  bias_s[tid] = bs[tid];
  bias_s[tid + 128] = bs[tid + 128];
  for (int e = tid; e < 16 * 64; e += blockDim.x) {
    const int b = e >> 6, hh = e & 63;
    const size_t si = (((size_t)seq * 16 + b) * 4 + l) * 64 + hh;
    h_s[e] = h0[si];
    c_s[e] = c0[si];
  }

  for (int t = 0; t < Tlen; ++t) {
    __syncthreads();  // prev step's h/c (and weight staging) complete
    if (wave == 0) {
      // TDM 2D gather of the 16-row x_t batch into LDS (row stride varies).
      const float* src;
      unsigned stride;
      if (today)      { src = shortp + (size_t)t * 128;               stride = 3840u; }
      else if (t < 8) { src = nmem + ((size_t)seq * 128 + t) * 128;   stride = 1024u; }
      else            { src = nq + (size_t)seq * 128;                 stride = 512u;  }
      tdm_load_2d_to_lds(src, (unsigned)(size_t)xt_s, 128u, 16u, stride);
      __builtin_amdgcn_s_wait_tensorcnt(0);
    }
    __syncthreads();

    v16bf ax[4], ah[2];
#pragma unroll
    for (int kk = 0; kk < 4; ++kk) ax[kk] = load_a_frag(xt_s, 128, kk * 32, lane);
#pragma unroll
    for (int kk = 0; kk < 2; ++kk) ah[kk] = load_a_frag(h_s, 64, kk * 32, lane);

#pragma unroll
    for (int nt = 0; nt < 4; ++nt) {
      const int n0 = wave * 64 + nt * 16;
      const int ft = (wave * 4 + nt);
      v8f c = {};
#pragma unroll
      for (int kk = 0; kk < 4; ++kk)
        c = WMMA_BF16(ax[kk], load_b_lds(wx_lds, ft * 4 + kk, lane), c);
#pragma unroll
      for (int kk = 0; kk < 2; ++kk)
        c = WMMA_BF16(ah[kk], load_b_lds(wh_lds, ft * 2 + kk, lane), c);
      const int col = n0 + (lane & 15);
      const int rbase = (lane >> 4) * 8;
#pragma unroll
      for (int j = 0; j < 8; ++j) g_s[(rbase + j) * 256 + col] = c[j] + bias_s[col];
    }
    __syncthreads();

    for (int e = tid; e < 16 * 64; e += blockDim.x) {
      const int b = e >> 6, hh = e & 63;
      const float* g = g_s + b * 256;
      const float ig = g[hh], fg = g[64 + hh], gg = g[128 + hh], og = g[192 + hh];
      const float c2 = sigm(fg) * c_s[e] + sigm(ig) * fast_tanh(gg);
      c_s[e] = c2;
      h_s[e] = sigm(og) * fast_tanh(c2);
    }
  }
  __syncthreads();
  const int colbase = today ? 1024 : seq * 256;
  for (int e = tid; e < 16 * 64; e += blockDim.x) {
    const int b = e >> 6, hh = e & 63;
    feats[(size_t)b * 1280 + colbase + l * 64 + hh] = h_s[e];
  }
}

// Kernel 11: final head.  y[b] = tanh(feats[b]@fcW + fcb) @ outW + outb
__global__ __launch_bounds__(128) void head_kernel(
    const float* __restrict__ feats, const float* __restrict__ fcW,
    const float* __restrict__ fcb, const float* __restrict__ outW,
    const float* __restrict__ outb, float* __restrict__ y) {
  __shared__ float yacc[16];
  const int j = threadIdx.x;  // 128 hidden cols
  if (j < 16) yacc[j] = 0.f;
  __syncthreads();
  for (int b = 0; b < 16; ++b) {
    float acc = 0.f;
    for (int kk = 0; kk < 1280; ++kk) acc += feats[(size_t)b * 1280 + kk] * fcW[(size_t)kk * 128 + j];
    atomicAdd(&yacc[b], fast_tanh(acc + fcb[j]) * outW[j]);
  }
  __syncthreads();
  if (j < 16) y[j] = yacc[j] + outb[0];
}

// ---------------------------------------------------------------------------
extern "C" void kernel_launch(void* const* d_in, const int* in_sizes, int n_in,
                              void* d_out, int out_size, void* d_ws, size_t ws_size,
                              hipStream_t stream) {
  const float* x     = (const float*)d_in[0];
  const float* mem   = (const float*)d_in[1];
  const float* th_W1 = (const float*)d_in[2];
  const float* th_b1 = (const float*)d_in[3];
  const float* th_W2 = (const float*)d_in[4];
  const float* th_b2 = (const float*)d_in[5];
  const float* sa_W1 = (const float*)d_in[6];
  const float* sa_b1 = (const float*)d_in[7];
  const float* sa_W2 = (const float*)d_in[8];
  const float* sa_b2 = (const float*)d_in[9];
  const float* bil   = (const float*)d_in[10];
  const float* rh_Wx = (const float*)d_in[11];
  const float* rh_Wh = (const float*)d_in[12];
  const float* rh_b  = (const float*)d_in[13];
  const float* rt_Wx = (const float*)d_in[14];
  const float* rt_Wh = (const float*)d_in[15];
  const float* rt_b  = (const float*)d_in[16];
  const float* fc_W  = (const float*)d_in[17];
  const float* fc_b  = (const float*)d_in[18];
  const float* out_W = (const float*)d_in[19];
  const float* out_b = (const float*)d_in[20];
  const float* n_st  = (const float*)d_in[21];
  const float* n_lt  = (const float*)d_in[22];
  const float* h0    = (const float*)d_in[23];
  const float* c0    = (const float*)d_in[24];
  const unsigned char* ltm_mask = (const unsigned char*)d_in[25];
  const unsigned char* stm_mask = (const unsigned char*)d_in[26];
  const float* sigma_p = (const float*)d_in[30];

  const int Nsamp = in_sizes[21] / (16 * 30);  // = 1000

  float* out = (float*)d_out;
  float* out_y    = out;                 // (16)
  float* out_ww   = out + 16;            // (16,30,20)
  float* out_wwm  = out + 16 + 9600;     // (16,256,20)
  float* out_stix = out + 91536;         // (16,4)   indices as float
  float* out_ltix = out + 91600;         // (16,4,8) indices as float

  float* ws = (float*)d_ws;
  float* sc_x   = ws;            // 9600
  float* sc_m   = ws + 9600;     // 81920
  float* shortp = ws + 91520;    // 16*30*128
  float* mem_e  = ws + 152960;   // 16*256*128
  float* sc_sa  = ws + 677248;   // 480
  float* qw     = ws + 677728;   // 480
  float* st_ind = ws + 678208;   // 16*4*30
  float* nq     = ws + 680128;   // 16*4*128
  float* tvec   = ws + 688320;   // 4*16*128
  float* wsoft  = ws + 696512;   // 4*16*256
  float* lt_ind = ws + 712896;   // 4*16*8*256
  float* nmem   = ws + 843968;   // 4*16*8*128
  float* feats  = ws + 909504;   // 16*1280

  // 1) token scores for x and mem (WMMA bf16, fused MLP, TDM-staged A tiles)
  mlp_score_kernel<<<(9600 / 16 + 7) / 8, 256, 0, stream>>>(x, 9600, th_W1, th_b1, th_W2, th_b2, sc_x);
  mlp_score_kernel<<<(81920 / 16 + 7) / 8, 256, 0, stream>>>(mem, 81920, th_W1, th_b1, th_W2, th_b2, sc_m);
  // 2) softmax over tokens + pooled embeddings (also emits ww / ww_mem)
  softmax_pool_kernel<<<(480 + 7) / 8, 256, 0, stream>>>(x, sc_x, out_ww, shortp, 480, 20);
  softmax_pool_kernel<<<(4096 + 7) / 8, 256, 0, stream>>>(mem, sc_m, out_wwm, mem_e, 4096, 20);
  // 3) event scores on pooled short rows, mask + softmax, exact top-4
  mlp_score_kernel<<<(480 / 16 + 7) / 8, 256, 0, stream>>>(shortp, 480, sa_W1, sa_b1, sa_W2, sa_b2, sc_sa);
  event_softmax_kernel<<<16, 32, 0, stream>>>(sc_sa, stm_mask, qw, out_stix);
  // 4) perturbed top-4 indicator + new queries
  ptopk_st_kernel<<<16, 256, 0, stream>>>(qw, n_st, sigma_p, st_ind, Nsamp);
  new_query_kernel<<<64, 128, 0, stream>>>(st_ind, shortp, nq);
  // 5) bilinear similarity + softmax over memory slots
  bilinear_t_kernel<<<64, 128, 0, stream>>>(bil, nq, tvec);
  sim_softmax_kernel<<<64, 256, 0, stream>>>(mem_e, tvec, ltm_mask, wsoft);
  // 6) perturbed top-8 indicator + retrieved memories
  ptopk_lt_kernel<<<64, 256, 0, stream>>>(wsoft, n_lt, sigma_p, lt_ind, out_ltix, Nsamp);
  new_mem_kernel<<<512, 128, 0, stream>>>(lt_ind, mem_e, nmem);
  // 7) RIM LSTM chains (WMMA gate GEMMs, LDS-resident weights) + head
  rim_kernel<<<20, 128, 0, stream>>>(shortp, nq, nmem, rh_Wx, rh_Wh, rh_b,
                                     rt_Wx, rt_Wh, rt_b, h0, c0, feats);
  head_kernel<<<1, 128, 0, stream>>>(feats, fc_W, fc_b, out_W, out_b, out_y);
}
